// ExternalAttention_83940840833098
// MI455X (gfx1250) — matrix operand order repaired and checked by reading, not compiled, not run on hardware
//
#include <hip/hip_runtime.h>
#include <math.h>

// ---------------------------------------------------------------------------
// Problem sizes (reference): B=16, H=W=64, C=512, K=64  -> tokens = 65536
// ---------------------------------------------------------------------------
#define TOKENS 65536
#define CDIM   512
#define KDIM   64
#define NTOK   4096   // tokens per batch (softmax axis)

typedef __attribute__((ext_vector_type(16))) __bf16        v16bf;
typedef __attribute__((ext_vector_type(8)))  float         v8f;
typedef __attribute__((ext_vector_type(4)))  float         v4f;
typedef __attribute__((ext_vector_type(4)))  unsigned int  v4u;
typedef __attribute__((ext_vector_type(2)))  unsigned int  v2u;

union Frag16 { v16bf bf; v4u u[2]; };

__device__ __forceinline__ unsigned short f32_to_bf16(float f) {
  unsigned int u = __builtin_bit_cast(unsigned int, f);
  unsigned int r = u + 0x7FFFu + ((u >> 16) & 1u);   // round-to-nearest-even
  return (unsigned short)(r >> 16);
}
__device__ __forceinline__ unsigned pack_bf16x2(float lo, float hi) {
  return (unsigned)f32_to_bf16(lo) | ((unsigned)f32_to_bf16(hi) << 16);
}

// Async global->LDS 16B copy (per lane). vdst = LDS byte offset (relative to
// the wave's LDS base == dynamic smem offset 0), vaddr = 64-bit global addr.
// Tracked with ASYNCcnt (ISA 10. / 15.18.3, opcode 98).
__device__ __forceinline__ void async_g2l_b128(unsigned lds_off, const void* g) {
  unsigned long long ga = (unsigned long long)g;
  asm volatile("global_load_async_to_lds_b128 %0, %1, off"
               :: "v"(lds_off), "v"(ga) : "memory");
}

// Workspace layout (bytes)
static constexpr size_t OFF_LOGITS = 0;                         // 16 MB
static constexpr size_t OFF_M      = 16u * 1024u * 1024u;       // 64KB bf16 M^T [64][512]
static constexpr size_t OFF_V      = OFF_M + 65536;             // 64KB bf16 V   [512][64]
static constexpr size_t OFF_B2     = OFF_V + 65536;             // 256B folded bias [64]
static constexpr size_t OFF_SC     = OFF_B2 + 256;              // 2KB  BN scale [512]
static constexpr size_t OFF_SH     = OFF_SC + 2048;             // 2KB  BN shift [512]
static constexpr size_t OFF_CM     = OFF_SH + 2048;             // 4KB  col max  [16][64]
static constexpr size_t OFF_CI     = OFF_CM + 4096;             // 4KB  col 1/sum[16][64]

// ---------------------------------------------------------------------------
// Kernel 0: fold weights.
//   M[c,k]  = sum_d w1[c,d]*m0[d,k]      stored as Mbf[k*512 + c]   (bf16)
//   V[k,d]  = sum_c m1[k,c]*w2[c,d]      stored as Vbf[d*64  + k]   (bf16)
//   bias2[k]= sum_d b1[d]*m0[d,k]
//   scale[d]= gamma[d]*rsqrt(var+eps); shift[d]= beta[d]-scale[d]*mean[d]
// ---------------------------------------------------------------------------
__global__ void __launch_bounds__(256) k_fold(
    const float* __restrict__ w1, const float* __restrict__ b1,
    const float* __restrict__ m0, const float* __restrict__ m1,
    const float* __restrict__ w2, const float* __restrict__ gamma,
    const float* __restrict__ beta, const float* __restrict__ mean,
    const float* __restrict__ var,
    unsigned short* __restrict__ Mbf, unsigned short* __restrict__ Vbf,
    float* __restrict__ bias2, float* __restrict__ scale,
    float* __restrict__ shift) {
  int i = blockIdx.x * blockDim.x + threadIdx.x;
  if (i >= 32768) return;
  {
    int k = i >> 9, c = i & 511;
    float acc = 0.f;
    for (int d = 0; d < CDIM; ++d) acc += w1[c * CDIM + d] * m0[d * KDIM + k];
    Mbf[i] = f32_to_bf16(acc);                    // flat = k*512 + c
  }
  {
    int d = i >> 6, kk = i & 63;
    float acc = 0.f;
    for (int c = 0; c < CDIM; ++c) acc += m1[kk * CDIM + c] * w2[c * CDIM + d];
    Vbf[i] = f32_to_bf16(acc);                    // flat = d*64 + kk
  }
  if (i < KDIM) {
    float acc = 0.f;
    for (int d = 0; d < CDIM; ++d) acc += b1[d] * m0[d * KDIM + i];
    bias2[i] = acc;
  }
  if (i < CDIM) {
    float s = gamma[i] * rsqrtf(var[i] + 1e-3f);
    scale[i] = s;
    shift[i] = beta[i] - s * mean[i];
  }
}

// ---------------------------------------------------------------------------
// Kernel 1: logits[65536,64] = bf16(X[65536,512]) @ M + bias2
// 256 threads (8 waves), 128-token tile. LDS: M^T 64KB (async preload) +
// double-buffered f32 X tiles (2 x 16KB, async, overlapped with WMMA).
// ---------------------------------------------------------------------------
#define G1_XOFF 65536u
#define G1_XBUF 16384u

__global__ void __launch_bounds__(256) k_gemm1(
    const float* __restrict__ X, const unsigned short* __restrict__ Mbf,
    const float* __restrict__ bias2, float* __restrict__ logits) {
  extern __shared__ char smem[];
  unsigned short* Mlds = (unsigned short*)smem;            // [64][512] bf16
  const int t = threadIdx.x;
  const int rbase = blockIdx.x * 128;

  // Async preload of whole folded M (64 KB): 16 x b128 per thread.
#pragma unroll
  for (int i = 0; i < 16; ++i) {
    unsigned e = (unsigned)(t + 256 * i) * 16u;
    async_g2l_b128(e, (const char*)Mbf + e);
  }
  // Async stage of X tile kb=0 into buffer 0 (f32, 128x32).
#pragma unroll
  for (int p = 0; p < 4; ++p) {
    int f = t + 256 * p;
    int row = f >> 3, c4 = (f & 7) * 4;
    async_g2l_b128(G1_XOFF + (unsigned)(row * 128 + c4 * 4),
                   X + (size_t)(rbase + row) * CDIM + c4);
  }

  const int w = t >> 5;        // wave id 0..7 -> rows 16w..16w+15
  const int lane = t & 31;
  const int m = lane & 15;
  const int h = lane >> 4;

  v8f acc[4] = {};

  for (int kb = 0; kb < 16; ++kb) {   // K-dim = C = 512 in steps of 32
    if (kb + 1 < 16) {
      // Issue async copy of next tile into the other buffer, then wait until
      // only those 4 copies remain outstanding (=> tile kb and M are in LDS).
      unsigned boff = G1_XOFF + ((unsigned)(kb + 1) & 1u) * G1_XBUF;
#pragma unroll
      for (int p = 0; p < 4; ++p) {
        int f = t + 256 * p;
        int row = f >> 3, c4 = (f & 7) * 4;
        async_g2l_b128(boff + (unsigned)(row * 128 + c4 * 4),
                       X + (size_t)(rbase + row) * CDIM + (kb + 1) * 32 + c4);
      }
      asm volatile("s_wait_asynccnt 0x4" ::: "memory");
    } else {
      asm volatile("s_wait_asynccnt 0x0" ::: "memory");
    }
    __syncthreads();

    // A fragment 16x32 bf16 (ISA 7.12.2): v<4 -> K=2v+8h, v>=4 -> K=16+2(v-4)+8h.
    // Read f32 from LDS (4 x ds_load_b128) and convert at frag build.
    const float* Arow =
        (const float*)(smem + G1_XOFF + ((unsigned)kb & 1u) * G1_XBUF) +
        (w * 16 + m) * 32;
    v4f f0 = *(const v4f*)(Arow + 8 * h);           // K = 8h .. 8h+3
    v4f f1 = *(const v4f*)(Arow + 8 * h + 4);       // K = 8h+4 .. 8h+7
    v4f f2 = *(const v4f*)(Arow + 16 + 8 * h);      // K = 16+8h ..
    v4f f3 = *(const v4f*)(Arow + 16 + 8 * h + 4);
    Frag16 a;
    a.u[0] = (v4u){pack_bf16x2(f0.x, f0.y), pack_bf16x2(f0.z, f0.w),
                   pack_bf16x2(f1.x, f1.y), pack_bf16x2(f1.z, f1.w)};
    a.u[1] = (v4u){pack_bf16x2(f2.x, f2.y), pack_bf16x2(f2.z, f2.w),
                   pack_bf16x2(f3.x, f3.y), pack_bf16x2(f3.z, f3.w)};

#pragma unroll
    for (int nt = 0; nt < 4; ++nt) {
      // B fragment: 32x16 bf16, column-per-lane: v<4 -> K=2v+8h, v>=4 -> K=16+2(v-4)+8h.
      Frag16 bfr;
      const char* Bb = (const char*)Mlds + (nt * 16 + m) * 1024 + kb * 64 + h * 16;
      bfr.u[0] = *(const v4u*)(Bb);
      bfr.u[1] = *(const v4u*)(Bb + 32);
      acc[nt] = __builtin_amdgcn_wmma_f32_16x16x32_bf16(
          false, a.bf, false, bfr.bf, (short)0, acc[nt], false, false);
    }
    __syncthreads();   // all reads of buffer kb&1 done before it is refilled
  }

  // Epilogue: + folded bias, store f32 logits. D layout: VGPR r -> row r+8h.
#pragma unroll
  for (int nt = 0; nt < 4; ++nt) {
    int col = nt * 16 + m;
    float bv = bias2[col];
#pragma unroll
    for (int r = 0; r < 8; ++r) {
      int row = rbase + w * 16 + r + 8 * h;
      logits[(size_t)row * KDIM + col] = acc[nt][r] + bv;
    }
  }
}

// ---------------------------------------------------------------------------
// Kernel 2: per-(batch, k) column softmax stats over N=4096 tokens.
// Online max/sum; stores max and 1/(sum*(1+1e-9)) (folds double-norm).
// ---------------------------------------------------------------------------
__global__ void __launch_bounds__(256) k_colred(
    const float* __restrict__ logits, float* __restrict__ cmax,
    float* __restrict__ cinv) {
  __shared__ float sm[4][64];
  __shared__ float ss[4][64];
  const int t = threadIdx.x;
  const int b = blockIdx.x;
  const int k = t & 63, g = t >> 6;
  const float* base = logits + (size_t)b * NTOK * KDIM + k;
  float mx = -__builtin_inff(), s = 0.f;
  for (int i = 0; i < 1024; ++i) {
    float v = base[(size_t)(g + 4 * i) * KDIM];
    float mn = fmaxf(mx, v);
    s = s * __expf(mx - mn) + __expf(v - mn);
    mx = mn;
  }
  sm[g][k] = mx;
  ss[g][k] = s;
  __syncthreads();
  if (t < 64) {
    float M = sm[0][t];
#pragma unroll
    for (int g2 = 1; g2 < 4; ++g2) M = fmaxf(M, sm[g2][t]);
    float S = 0.f;
#pragma unroll
    for (int g2 = 0; g2 < 4; ++g2) S += ss[g2][t] * __expf(sm[g2][t] - M);
    cmax[b * 64 + t] = M;
    cinv[b * 64 + t] = 1.0f / (S * (1.0f + 1e-9f));
  }
}

// ---------------------------------------------------------------------------
// Kernel 3: out = relu(scale * (attn @ V) + shift + X)
// V slice arrives via async copy overlapped with the exp/softmax staging.
// Tile: 128 tokens x 128 output channels; K = 64 -> two 32-K WMMA steps.
// ---------------------------------------------------------------------------
__global__ void __launch_bounds__(256) k_gemm2(
    const float* __restrict__ X, const float* __restrict__ logits,
    const unsigned short* __restrict__ Vbf, const float* __restrict__ cmax,
    const float* __restrict__ cinv, const float* __restrict__ scale,
    const float* __restrict__ shift, float* __restrict__ out) {
  extern __shared__ char smem[];
  unsigned short* Vlds = (unsigned short*)smem;             // [128][64] bf16
  unsigned short* Alds = (unsigned short*)(smem + 16384);   // [128][64] bf16
  float* cmL = (float*)(smem + 32768);                      // [64]
  float* ciL = (float*)(smem + 33024);                      // [64]
  const int t = threadIdx.x;
  const int rbase = blockIdx.x * 128;   // token tile (within one batch)
  const int ct = blockIdx.y;            // channel tile (128 channels)
  const int b = rbase >> 12;            // rbase / 4096

  // Async preload of V slice [128 ch][64 k] bf16 (16 KB): 4 x b128 per thread.
#pragma unroll
  for (int i = 0; i < 4; ++i) {
    unsigned e = (unsigned)(t + 256 * i) * 16u;
    async_g2l_b128(e, (const char*)(Vbf + (size_t)ct * 128 * KDIM) + e);
  }
  if (t < 64) {
    cmL[t] = cmax[b * 64 + t];
    ciL[t] = cinv[b * 64 + t];
  }
  __syncthreads();   // cmL/ciL visible (V copy still in flight)

  // Stage attn tile: exp(logit - max) * inv -> bf16 LDS (overlaps V copy).
#pragma unroll
  for (int p = 0; p < 8; ++p) {
    int f = t + 256 * p;
    int row = f >> 4, c4 = (f & 15) * 4;
    v4f v = *(const v4f*)(logits + (size_t)(rbase + row) * KDIM + c4);
    float a0 = __expf(v.x - cmL[c4 + 0]) * ciL[c4 + 0];
    float a1 = __expf(v.y - cmL[c4 + 1]) * ciL[c4 + 1];
    float a2 = __expf(v.z - cmL[c4 + 2]) * ciL[c4 + 2];
    float a3 = __expf(v.w - cmL[c4 + 3]) * ciL[c4 + 3];
    v2u pk;
    pk.x = pack_bf16x2(a0, a1);
    pk.y = pack_bf16x2(a2, a3);
    *(v2u*)(Alds + row * 64 + c4) = pk;
  }
  asm volatile("s_wait_asynccnt 0x0" ::: "memory");   // V slice landed
  __syncthreads();

  const int w = t >> 5, lane = t & 31, m = lane & 15, h = lane >> 4;
  v8f acc[8] = {};
#pragma unroll
  for (int ks = 0; ks < 2; ++ks) {
    Frag16 a;
    const char* Ab = (const char*)Alds + (w * 16 + m) * 128 + ks * 64 + h * 16;
    a.u[0] = *(const v4u*)(Ab);
    a.u[1] = *(const v4u*)(Ab + 32);
#pragma unroll
    for (int nt = 0; nt < 8; ++nt) {
      Frag16 bfr;
      const char* Bb = (const char*)Vlds + (nt * 16 + m) * 128 + ks * 64 + h * 16;
      bfr.u[0] = *(const v4u*)(Bb);
      bfr.u[1] = *(const v4u*)(Bb + 32);
      acc[nt] = __builtin_amdgcn_wmma_f32_16x16x32_bf16(
          false, a.bf, false, bfr.bf, (short)0, acc[nt], false, false);
    }
  }

  // Fused epilogue: BN affine + residual + ReLU, f32 out.
#pragma unroll
  for (int nt = 0; nt < 8; ++nt) {
    int col = ct * 128 + nt * 16 + m;
    float sc = scale[col], sh = shift[col];
#pragma unroll
    for (int r = 0; r < 8; ++r) {
      int row = rbase + w * 16 + r + 8 * h;
      float xin = X[(size_t)row * CDIM + col];
      float o = sc * acc[nt][r] + sh + xin;
      out[(size_t)row * CDIM + col] = fmaxf(o, 0.f);
    }
  }
}

// ---------------------------------------------------------------------------
extern "C" void kernel_launch(void* const* d_in, const int* in_sizes, int n_in,
                              void* d_out, int out_size, void* d_ws,
                              size_t ws_size, hipStream_t stream) {
  const float* X     = (const float*)d_in[0];
  const float* w1    = (const float*)d_in[1];
  const float* b1    = (const float*)d_in[2];
  const float* m0    = (const float*)d_in[3];
  const float* m1    = (const float*)d_in[4];
  const float* w2    = (const float*)d_in[5];
  const float* gamma = (const float*)d_in[6];
  const float* beta  = (const float*)d_in[7];
  const float* mean  = (const float*)d_in[8];
  const float* var   = (const float*)d_in[9];
  float* out = (float*)d_out;

  char* ws = (char*)d_ws;
  float* logits        = (float*)(ws + OFF_LOGITS);
  unsigned short* Mbf  = (unsigned short*)(ws + OFF_M);
  unsigned short* Vbf  = (unsigned short*)(ws + OFF_V);
  float* bias2         = (float*)(ws + OFF_B2);
  float* scaleB        = (float*)(ws + OFF_SC);
  float* shiftB        = (float*)(ws + OFF_SH);
  float* cmaxB         = (float*)(ws + OFF_CM);
  float* cinvB         = (float*)(ws + OFF_CI);

  k_fold<<<128, 256, 0, stream>>>(w1, b1, m0, m1, w2, gamma, beta, mean, var,
                                  Mbf, Vbf, bias2, scaleB, shiftB);
  // LDS: 64KB M + 2x16KB double-buffered f32 X tiles
  k_gemm1<<<TOKENS / 128, 256, 98304, stream>>>(X, Mbf, bias2, logits);
  k_colred<<<16, 256, 0, stream>>>(logits, cmaxB, cinvB);
  // LDS: 16KB V + 16KB attn + 512B softmax stats
  k_gemm2<<<dim3(TOKENS / 128, CDIM / 128), 256, 33280, stream>>>(
      X, logits, Vbf, cmaxB, cinvB, scaleB, shiftB, out);
}
